// TNLayer_9440338117219
// MI455X (gfx1250) — matrix operand order repaired and checked by reading, not compiled, not run on hardware
//
#include <hip/hip_runtime.h>

typedef _Float16 f16;
typedef __attribute__((ext_vector_type(16))) _Float16 v16h;
typedef __attribute__((ext_vector_type(8)))  _Float16 v8h;
typedef __attribute__((ext_vector_type(8)))  float    v8f;

constexpr int kB = 4096;   // samples
constexpr int kM = 128;
constexpr int kJ = 64;
constexpr int kL = 64;
constexpr int kN = 128;
constexpr int kK = 4;
constexpr int S_PER_WG = 8;     // samples per workgroup
constexpr int P64  = 72;        // padded pitch for 64-wide f16 rows (144B, 16B aligned, bank-spread)
constexpr int P256 = 264;       // padded pitch for 256-wide f16 rows (528B, 16B aligned, bank-spread)

// ---- WMMA fragment loaders from K-major f16 LDS arrays (wave32 layouts, ISA 7.12.2) ----
// A matrix 16x32 f16: lane = (kk/8 within 16-group)*16 + m ; VGPR d: d0..3 -> K=g*16+h*8+{0..7},
// d4..7 -> K=g*16+16.. wait per ISA: V0..3 = K 0..15 half, V4..7 = K 16..31 half.
__device__ __forceinline__ v16h frag_a(const f16* __restrict__ base, int pitch, int kb, int lane) {
    const int row = lane & 15;
    const int kh  = lane >> 4;                 // which 8-K subgroup within each 16-K group
    const f16* p = base + row * pitch + kb + kh * 8;
    v8h lo = *(const v8h*)p;                   // K = kb + kh*8 + 0..7      -> VGPR 0..3
    v8h hi = *(const v8h*)(p + 16);            // K = kb + 16 + kh*8 + 0..7 -> VGPR 4..7
    v16h r;
#pragma unroll
    for (int i = 0; i < 8; ++i) { r[i] = lo[i]; r[i + 8] = hi[i]; }
    return r;
}
// B matrix 32x16 f16: same K order as A: lane half kg holds K=kg*8+0..7 and 16+kg*8+0..7 (n = lane&15).
__device__ __forceinline__ v16h frag_b(const f16* __restrict__ base, int pitch, int kb, int lane) {
    const int row = lane & 15;                 // n within tile
    const int kg  = lane >> 4;                 // which 8-K subgroup within each 16-K group
    const f16* p = base + row * pitch + kb + kg * 8;
    v8h lo = *(const v8h*)p;                   // K = kb + kg*8 + 0..7      -> VGPR 0..3
    v8h hi = *(const v8h*)(p + 16);            // K = kb + 16 + kg*8 + 0..7 -> VGPR 4..7
    v16h r;
#pragma unroll
    for (int i = 0; i < 8; ++i) { r[i] = lo[i]; r[i + 8] = hi[i]; }
    return r;
}

__global__ __launch_bounds__(256, 1)
void tn_layer_kernel(const float* __restrict__ x_all,   // (B, J*L)
                     const float* __restrict__ a,       // (M, J, K)
                     const float* __restrict__ c,       // (N, L, K)
                     const float* __restrict__ bias,    // (M, N)
                     float* __restrict__ out)           // (B, M*N)
{
    __shared__ __align__(16) f16 sA[kK * kM * P64];   // [k][m][j]   K-major over j
    __shared__ __align__(16) f16 sC[kN * P256];       // [n][k*64+l] K-major over (k,l)
    __shared__ __align__(16) f16 sXT[kL * P64];       // [l][j]      K-major over j
    __shared__ __align__(16) f16 sT[kM * P256];       // [m][k*64+l] K-major over (k,l)

    const int tid  = threadIdx.x;
    const int wave = tid >> 5;
    const int lane = tid & 31;

    // ---- one-time per WG: convert a, c to f16 LDS in WMMA-friendly K-major layouts ----
    for (int i = tid; i < kM * kJ * kK; i += 256) {      // a[m,j,k] -> sA[k][m][j]
        int m = i >> 8, j = (i >> 2) & 63, k = i & 3;
        sA[(k * kM + m) * P64 + j] = (f16)a[i];
    }
    for (int i = tid; i < kN * kL * kK; i += 256) {      // c[n,l,k] -> sC[n][k*64+l]
        int n = i >> 8, l = (i >> 2) & 63, k = i & 3;
        sC[n * P256 + k * 64 + l] = (f16)c[i];
    }

    // ---- bias tile for this wave's output row block (mt == wave), kept in VGPRs ----
    const int mrow = wave * 16 + ((lane >> 4) << 3);     // C/D layout: m = r + (lane>=16)*8
    const int ncol = lane & 15;
    v8f breg[8];
#pragma unroll
    for (int nt = 0; nt < 8; ++nt)
#pragma unroll
        for (int r = 0; r < 8; ++r)
            breg[nt][r] = bias[(mrow + r) * kN + nt * 16 + ncol];

    __syncthreads();

    const int b0 = blockIdx.x * S_PER_WG;
    for (int s = 0; s < S_PER_WG; ++s) {
        const int b = b0 + s;
        const float* xb = x_all + (size_t)b * (kJ * kL);

        // x[j,l] -> sXT[l][j], f16, streaming (read-once) loads
        for (int i = tid; i < kJ * kL; i += 256) {
            int j = i >> 6, l = i & 63;
            sXT[l * P64 + j] = (f16)__builtin_nontemporal_load(&xb[i]);
        }
        // prefetch next sample's activations while we compute this one
        if (s + 1 < S_PER_WG) {
            const float* xn = x_all + (size_t)(b + 1) * (kJ * kL);
            __builtin_prefetch(xn + tid * 16, 0, 0);
        }
        __syncthreads();

        // ---- Stage 1: for each k, T_k(128x64) = A_k(128x64) @ X(64x64); 128 tiles, 2 wmma each ----
        for (int i = 0; i < 16; ++i) {
            const int t  = wave * 16 + i;
            const int k  = t >> 5;
            const int mt = (t >> 2) & 7;
            const int lt = t & 3;
            const f16* Ab = &sA[(k * kM + mt * 16) * P64];
            const f16* Bb = &sXT[(lt * 16) * P64];
            v8f acc = {};
            acc = __builtin_amdgcn_wmma_f32_16x16x32_f16(
                false, frag_a(Ab, P64, 0, lane), false, frag_b(Bb, P64, 0, lane),
                (short)0, acc, false, false);
            acc = __builtin_amdgcn_wmma_f32_16x16x32_f16(
                false, frag_a(Ab, P64, 32, lane), false, frag_b(Bb, P64, 32, lane),
                (short)0, acc, false, false);
            // D tile -> sT[m][k*64 + lt*16 + n_local] as f16
            const int col   = k * 64 + lt * 16 + (lane & 15);
            const int mbase = mt * 16 + ((lane >> 4) << 3);
#pragma unroll
            for (int r = 0; r < 8; ++r)
                sT[(mbase + r) * P256 + col] = (f16)acc[r];
        }
        __syncthreads();

        // ---- Stage 2: out(128x128) = T(128x256) @ Chat^T(256x128) + bias ----
        // wave owns m-tile == wave; hoist its 8 A-fragments across all 8 n-tiles
        v16h afr[8];
        const f16* Tb = &sT[(wave * 16) * P256];
#pragma unroll
        for (int kb = 0; kb < 8; ++kb) afr[kb] = frag_a(Tb, P256, kb * 32, lane);

        float* ob = out + (size_t)b * (kM * kN);
#pragma unroll
        for (int nt = 0; nt < 8; ++nt) {
            const f16* Cb = &sC[(nt * 16) * P256];
            v8f acc = {};
#pragma unroll
            for (int kb = 0; kb < 8; ++kb)
                acc = __builtin_amdgcn_wmma_f32_16x16x32_f16(
                    false, afr[kb], false, frag_b(Cb, P256, kb * 32, lane),
                    (short)0, acc, false, false);
#pragma unroll
            for (int r = 0; r < 8; ++r) {
                float v = acc[r] + breg[nt][r];
                __builtin_nontemporal_store(v, &ob[(mrow + r) * kN + nt * 16 + ncol]);
            }
        }
        __syncthreads();  // protect sXT/sT reuse next sample
    }
}

extern "C" void kernel_launch(void* const* d_in, const int* in_sizes, int n_in,
                              void* d_out, int out_size, void* d_ws, size_t ws_size,
                              hipStream_t stream) {
    const float* x    = (const float*)d_in[0];
    const float* a    = (const float*)d_in[1];
    const float* c    = (const float*)d_in[2];
    const float* bias = (const float*)d_in[3];
    float* out = (float*)d_out;

    dim3 grid(kB / S_PER_WG);   // 512 workgroups
    dim3 block(256);            // 8 wave32 waves
    tn_layer_kernel<<<grid, block, 0, stream>>>(x, a, c, bias, out);
}
